// MultiHeadAttention_88819923682032
// MI455X (gfx1250) — matrix operand order repaired and checked by reading, not compiled, not run on hardware
//
#include <hip/hip_runtime.h>
#include <hip/hip_bf16.h>

// ---- problem constants (from reference) ----
#define BATCH 2
#define SEQ   2048
#define DMODEL 1024
#define NHEAD 16
#define HDIM  64      // DMODEL / NHEAD
#define MROWS (BATCH * SEQ)          // 4096
#define ACT_ELEMS ((size_t)BATCH * SEQ * DMODEL)   // 4194304
#define W_ELEMS   ((size_t)DMODEL * DMODEL)        // 1048576

typedef __attribute__((ext_vector_type(16))) __bf16 v16bf;
typedef __attribute__((ext_vector_type(8)))  __bf16 v8bf;
typedef __attribute__((ext_vector_type(8)))  float  v8f;
typedef __attribute__((ext_vector_type(4)))  float  v4f;
typedef __attribute__((ext_vector_type(4)))  unsigned int v4u;
typedef __attribute__((ext_vector_type(8)))  unsigned int v8u;

static __device__ __forceinline__ v16bf combine16(v8bf lo, v8bf hi) {
    v16bf r;
#pragma unroll
    for (int i = 0; i < 8; ++i) { r[i] = lo[i]; r[i + 8] = hi[i]; }
    return r;
}

static __device__ __forceinline__ v8bf ld8(const __bf16* p) {
    return *(const v8bf*)p;
}

// A-fragment (16x32 bf16): lane hi-half selects k-chunks {hi*8..+7, 16+hi*8..+7}
static __device__ __forceinline__ v16bf frag_a(const __bf16* rowbase, int hi) {
    return combine16(ld8(rowbase + hi * 8), ld8(rowbase + 16 + hi * 8));
}
// B-fragment (32x16 bf16): lane hi-half selects k-chunks {hi*8..+7, 16+hi*8..+7}
static __device__ __forceinline__ v16bf frag_b(const __bf16* rowbase, int hi) {
    return combine16(ld8(rowbase + hi * 8), ld8(rowbase + 16 + hi * 8));
}

static __device__ __forceinline__ v8f wmma_bf16(v16bf a, v16bf b, v8f c) {
    return __builtin_amdgcn_wmma_f32_16x16x32_bf16(
        false, a, false, b, (short)0, c, false, false);
}

// ---------------- Tensor Data Mover descriptors (CDNA5 ISA ch.8) ----------------
// Group 0: [1:0]=count=1, [63:32]=lds_addr, [120:64]=global_addr, [127:126]=type=2
static __device__ __forceinline__ v4u tdm_g0(unsigned lds_addr, unsigned long long ga) {
    v4u g;
    g.x = 1u;
    g.y = lds_addr;
    g.z = (unsigned)(ga & 0xffffffffull);
    g.w = (unsigned)((ga >> 32) & 0x1ffffffull) | (2u << 30);
    return g;
}
// Group 1 for a 2-D bf16 tile (data_size=1 -> 2B units), no multicast,
// optional LDS row padding (interval/amount are encoded codes per D# spec).
static __device__ __forceinline__ v8u tdm_g1(unsigned tensor_d0, unsigned tensor_d1,
                                             unsigned tile_d0, unsigned tile_d1,
                                             unsigned long long d0_stride,
                                             unsigned pad_en, unsigned pad_ivl,
                                             unsigned pad_amt) {
    v8u g;
    g[0] = (1u << 16) | (pad_en << 20) | (pad_ivl << 22) | (pad_amt << 25);
    g[1] = (tensor_d0 & 0xffffu) << 16;                       // dim0[15:0] @ [63:48]
    g[2] = ((tensor_d0 >> 16) & 0xffffu) | ((tensor_d1 & 0xffffu) << 16);
    g[3] = ((tensor_d1 >> 16) & 0xffffu) | ((tile_d0 & 0xffffu) << 16);
    g[4] = (tile_d1 & 0xffffu);                               // tile_dim2 = 0
    g[5] = (unsigned)(d0_stride & 0xffffffffull);
    g[6] = (unsigned)((d0_stride >> 32) & 0xffffull);         // dim1_stride = 0 (2D)
    g[7] = 0u;
    return g;
}
// 2-D form: groups 2/3 are NULL. Inline asm is portable across both toolchains
// (the clang builtin arity differs between ROCm 7.2 and amdgpu-toolchain).
static __device__ __forceinline__ void tdm_load2d(v4u g0, v8u g1) {
    asm volatile("tensor_load_to_lds %0, %1" :: "s"(g0), "s"(g1) : "memory");
}

// ---------------- fp32 -> bf16 cast (4 elems / thread) ----------------
__global__ __launch_bounds__(256) void cast_bf16_kernel(const float* __restrict__ src,
                                                        __bf16* __restrict__ dst, int n4) {
    int i = (blockIdx.x * 256 + threadIdx.x);
    if (i < n4) {
        v4f x = *(const v4f*)&src[i * 4];
        __bf16 y[4];
#pragma unroll
        for (int e = 0; e < 4; ++e) y[e] = (__bf16)x[e];
        *(ulong1*)&dst[i * 4] = *(ulong1*)y;
    }
}

// ---------------- tiled WMMA GEMM:  C[m,n] = (sum_k A[m,k]*W[n,k] + bias[n]) * scale ----
// A: [M,K] bf16 row-major.  W: [N,K] bf16 row-major (torch Linear weight layout).
// TDM-staged, double-buffered LDS; 64x128x64 tiles; 8 waves each own a 32x32 patch.
#define GBM 64
#define GBN 128
#define GBK 64
#define GBKP (GBK + 8)   // padded LDS row stride (bf16) == TDM pad 16B per 128B row

template <bool OUT_BF16>
__global__ __launch_bounds__(256) void gemm_xwT_wmma(
    const __bf16* __restrict__ A, const __bf16* __restrict__ W,
    const float* __restrict__ bias, void* __restrict__ Cout,
    int M, int N, int K, float scale)
{
    __shared__ __bf16 As[2][GBM * GBKP];   // 2 x 9 KB
    __shared__ __bf16 Bs[2][GBN * GBKP];   // 2 x 18 KB

    const int tid  = threadIdx.x;
    const int lane = tid & 31;
    const int wv   = tid >> 5;          // 8 waves
    const int lr   = lane & 15;
    const int hi   = lane >> 4;
    const int m0   = blockIdx.x * GBM;
    const int n0   = blockIdx.y * GBN;
    const int wm   = wv & 1;            // 2 m-subtiles of 32
    const int wn   = wv >> 1;           // 4 n-subtiles of 32

    // TDM setup: pad_ivl code 4 => every 32 DWORDs (one 128B row),
    // pad_amt code 3 => 4 DWORDs (16B)  ->  LDS row stride GBKP.
    const unsigned ldsA[2] = { (unsigned)(size_t)&As[0][0], (unsigned)(size_t)&As[1][0] };
    const unsigned ldsB[2] = { (unsigned)(size_t)&Bs[0][0], (unsigned)(size_t)&Bs[1][0] };
    const unsigned long long aBase =
        (unsigned long long)(size_t)A + (unsigned long long)(size_t)m0 * K * 2ull;
    const unsigned long long bBase =
        (unsigned long long)(size_t)W + (unsigned long long)(size_t)n0 * K * 2ull;
    const v8u g1A = tdm_g1((unsigned)K, GBM, GBK, GBM, (unsigned long long)K, 1u, 4u, 3u);
    const v8u g1B = tdm_g1((unsigned)K, GBN, GBK, GBN, (unsigned long long)K, 1u, 4u, 3u);

    if (wv == 0) {                      // prologue DMA for tile 0
        tdm_load2d(tdm_g0(ldsA[0], aBase), g1A);
        tdm_load2d(tdm_g0(ldsB[0], bBase), g1B);
        __builtin_amdgcn_s_wait_tensorcnt(0);
    }
    __syncthreads();

    v8f acc[2][2] = {};
    int cur = 0;
    for (int k0 = 0; k0 < K; k0 += GBK) {
        const bool has_next = (k0 + GBK) < K;
        if (wv == 0 && has_next) {      // async DMA of next tile into alt buffer
            const unsigned long long koff = (unsigned long long)(k0 + GBK) * 2ull;
            tdm_load2d(tdm_g0(ldsA[cur ^ 1], aBase + koff), g1A);
            tdm_load2d(tdm_g0(ldsB[cur ^ 1], bBase + koff), g1B);
        }
#pragma unroll
        for (int kk = 0; kk < GBK; kk += 32) {
            v16bf af[2], bf[2];
#pragma unroll
            for (int i = 0; i < 2; ++i)
                af[i] = frag_a(&As[cur][(wm * 32 + i * 16 + lr) * GBKP + kk], hi);
#pragma unroll
            for (int j = 0; j < 2; ++j)
                bf[j] = frag_b(&Bs[cur][(wn * 32 + j * 16 + lr) * GBKP + kk], hi);
#pragma unroll
            for (int i = 0; i < 2; ++i)
#pragma unroll
                for (int j = 0; j < 2; ++j)
                    acc[i][j] = wmma_bf16(af[i], bf[j], acc[i][j]);
        }
        if (wv == 0 && has_next) __builtin_amdgcn_s_wait_tensorcnt(0);
        __syncthreads();                // publish DMA'd tile; retire reads of old one
        cur ^= 1;
    }

    // epilogue (branchless; row = hi*8 + e, col = lr within each 16x16 tile)
    float*  Cf = (float*)Cout;
    __bf16* Cb = (__bf16*)Cout;
#pragma unroll
    for (int i = 0; i < 2; ++i) {
#pragma unroll
        for (int j = 0; j < 2; ++j) {
            const int mt = m0 + wm * 32 + i * 16 + hi * 8;
            const int nt = n0 + wn * 32 + j * 16 + lr;
            const float bnv = bias[nt];
#pragma unroll
            for (int e = 0; e < 8; ++e) {
                const float vvv = (acc[i][j][e] + bnv) * scale;
                const size_t idx = (size_t)(mt + e) * N + nt;
                if constexpr (OUT_BF16) Cb[idx] = (__bf16)vvv;
                else                    Cf[idx] = vvv;
            }
        }
    }
}

// ---------------- fused attention: scores -> softmax(+weights out) -> P@V ----------------
// One block handles (b, h, 32 query rows). 256 threads = 8 waves.
// LDS: sS = 32x2048 bf16 scores/probs (128KB), sVt = 64x256 bf16 V^T chunk (32KB),
//      sRed = 256 floats reduction scratch. ~165KB of the 320KB WGP LDS.
#define QT  32
#define VCH 256
#define ATTN_LDS (QT * SEQ * 2 + HDIM * VCH * 2 + 256 * 4)

__global__ __launch_bounds__(256) void attn_fused_wmma(
    const __bf16* __restrict__ Q, const __bf16* __restrict__ Kx,
    const __bf16* __restrict__ V, const int* __restrict__ mask,
    float* __restrict__ wout /* [B,H,S,S] fp32 */,
    __bf16* __restrict__ ctx /* [B,S,D] bf16 */)
{
    extern __shared__ char smem[];
    __bf16* sS  = (__bf16*)smem;                              // QT x SEQ
    __bf16* sVt = (__bf16*)(smem + (size_t)QT * SEQ * 2);     // HDIM x VCH
    float*  sRed = (float*)(smem + (size_t)QT * SEQ * 2 + (size_t)HDIM * VCH * 2);

    const int tid  = threadIdx.x;
    const int lane = tid & 31;
    const int wv   = tid >> 5;
    const int lr   = lane & 15;
    const int hi   = lane >> 4;

    const int nqt = SEQ / QT;                 // 64
    const int qt  = blockIdx.x % nqt;
    const int h   = (blockIdx.x / nqt) % NHEAD;
    const int b   = blockIdx.x / (nqt * NHEAD);

    const int qsub  = wv & 1;                 // 2 q-subtiles of 16 rows
    const int strip = wv >> 1;                // 4 n-strips of 512 cols

    // ---- Q A-fragments pinned in registers (k = 0..63 in two 32-chunks) ----
    const size_t qrow = (size_t)(b * SEQ + qt * QT + qsub * 16 + lr) * DMODEL + h * HDIM;
    v16bf aq[2];
#pragma unroll
    for (int kk = 0; kk < 2; ++kk)
        aq[kk] = frag_a(&Q[qrow + kk * 32], hi);

    // ---- scores: 16 x 512 strip per wave, B-fragments streamed from global K ----
    for (int t = 0; t < 32; ++t) {
        const int n0 = strip * 512 + t * 16;
        const size_t krow = (size_t)(b * SEQ + n0 + lr) * DMODEL + h * HDIM;
        if (t + 1 < 32) {   // prefetch next K tile row (global_prefetch_b8)
            __builtin_prefetch(&Kx[krow + (size_t)16 * DMODEL], 0, 1);
        }
        v8f acc = {};
#pragma unroll
        for (int kk = 0; kk < 2; ++kk)
            acc = wmma_bf16(aq[kk], frag_b(&Kx[krow + kk * 32], hi), acc);
        // scale folded into Q projection; branchless mask; stash as bf16
        const int n = n0 + lr;
        const float maskbias = (mask[b * SEQ + n] == 0) ? -1e10f : 0.0f;
#pragma unroll
        for (int e = 0; e < 8; ++e) {
            const int ql = qsub * 16 + hi * 8 + e;
            sS[ql * SEQ + n] = (__bf16)(acc[e] + maskbias);
        }
    }
    __syncthreads();

    // ---- softmax: 8 threads per row; interleaved chunks so a row's 8 threads
    //      cover 256B-contiguous global runs when writing the fp32 weights ----
    {
        const int r   = tid >> 3;             // 0..31
        const int seg = tid & 7;
        float mx = -3.0e38f;
        for (int i = 0; i < SEQ / 64; ++i) {
            const int n = i * 64 + seg * 8;
            v8bf x = *(v8bf*)&sS[r * SEQ + n];
#pragma unroll
            for (int e = 0; e < 8; ++e) mx = fmaxf(mx, (float)x[e]);
        }
        sRed[tid] = mx;
        __syncthreads();
        float rowmax = -3.0e38f;
#pragma unroll
        for (int i = 0; i < 8; ++i) rowmax = fmaxf(rowmax, sRed[r * 8 + i]);
        __syncthreads();
        float sm = 0.f;
        for (int i = 0; i < SEQ / 64; ++i) {
            const int n = i * 64 + seg * 8;
            v8bf x = *(v8bf*)&sS[r * SEQ + n];
#pragma unroll
            for (int e = 0; e < 8; ++e) sm += __expf((float)x[e] - rowmax);
        }
        sRed[tid] = sm;
        __syncthreads();
        float rowsum = 0.f;
#pragma unroll
        for (int i = 0; i < 8; ++i) rowsum += sRed[r * 8 + i];
        const float inv = 1.0f / rowsum;
        float* wrow = wout + ((size_t)(b * NHEAD + h) * SEQ + qt * QT + r) * SEQ;
        for (int i = 0; i < SEQ / 64; ++i) {
            const int n = i * 64 + seg * 8;
            v8bf x = *(v8bf*)&sS[r * SEQ + n];
            v8f pv;
#pragma unroll
            for (int e = 0; e < 8; ++e) {
                const float p = __expf((float)x[e] - rowmax) * inv;
                pv[e] = p;
                x[e] = (__bf16)p;
            }
            *(v8f*)&wrow[n] = pv;              // 32B vector store, coalesced runs
            *(v8bf*)&sS[r * SEQ + n] = x;
        }
    }
    __syncthreads();

    // ---- ctx = P @ V : one 16x16 output tile per wave (2 qsub x 4 d-tiles) ----
    const int dt = wv >> 1;                   // 4 d-tiles of 16
    v8f cacc = {};
    for (int ch = 0; ch < SEQ; ch += VCH) {
        // stage V^T chunk: thread t owns source row n = ch + t
        {
            const size_t vrow = (size_t)(b * SEQ + ch + tid) * DMODEL + h * HDIM;
            if (ch + VCH < SEQ)
                __builtin_prefetch(&V[vrow + (size_t)VCH * DMODEL], 0, 1);
#pragma unroll
            for (int dd = 0; dd < HDIM; dd += 8) {
                v8bf vvv = ld8(&V[vrow + dd]);
#pragma unroll
                for (int e = 0; e < 8; ++e) sVt[(dd + e) * VCH + tid] = vvv[e];
            }
        }
        __syncthreads();
#pragma unroll
        for (int kk = 0; kk < VCH; kk += 32) {
            v16bf af = frag_a(&sS[(qsub * 16 + lr) * SEQ + ch + kk], hi);
            v16bf bf = frag_b(&sVt[(dt * 16 + lr) * VCH + kk], hi);
            cacc = wmma_bf16(af, bf, cacc);
        }
        __syncthreads();
    }
    {
        const int q = qt * QT + qsub * 16 + hi * 8;
        const int d = dt * 16 + lr;
#pragma unroll
        for (int e = 0; e < 8; ++e)
            ctx[(size_t)(b * SEQ + q + e) * DMODEL + h * HDIM + d] = (__bf16)cacc[e];
    }
}

// ---------------- host-side launch ----------------
extern "C" void kernel_launch(void* const* d_in, const int* in_sizes, int n_in,
                              void* d_out, int out_size, void* d_ws, size_t ws_size,
                              hipStream_t stream) {
    const float* q_in = (const float*)d_in[0];
    const float* k_in = (const float*)d_in[1];
    const float* v_in = (const float*)d_in[2];
    const int*   mask = (const int*)d_in[3];
    const float* Wq = (const float*)d_in[4];  const float* bq = (const float*)d_in[5];
    const float* Wk = (const float*)d_in[6];  const float* bk = (const float*)d_in[7];
    const float* Wv = (const float*)d_in[8];  const float* bv = (const float*)d_in[9];
    const float* Wo = (const float*)d_in[10]; const float* bo = (const float*)d_in[11];

    float* out = (float*)d_out;                 // [B,S,D] fp32
    float* weights_out = out + ACT_ELEMS;       // [B,H,S,S] fp32

    // workspace carve-up (all bf16): 3 act casts, 4 weight casts, Q/K/V, ctx = 64 MB
    char* p = (char*)d_ws;
    __bf16* qx  = (__bf16*)p; p += ACT_ELEMS * 2;
    __bf16* kx  = (__bf16*)p; p += ACT_ELEMS * 2;
    __bf16* vx  = (__bf16*)p; p += ACT_ELEMS * 2;
    __bf16* wqx = (__bf16*)p; p += W_ELEMS * 2;
    __bf16* wkx = (__bf16*)p; p += W_ELEMS * 2;
    __bf16* wvx = (__bf16*)p; p += W_ELEMS * 2;
    __bf16* wox = (__bf16*)p; p += W_ELEMS * 2;
    __bf16* Qp  = (__bf16*)p; p += ACT_ELEMS * 2;
    __bf16* Kp  = (__bf16*)p; p += ACT_ELEMS * 2;
    __bf16* Vp  = (__bf16*)p; p += ACT_ELEMS * 2;
    __bf16* Cx  = (__bf16*)p; p += ACT_ELEMS * 2;

    // 1) fp32 -> bf16 casts (vectorized x4)
    {
        int na4 = (int)(ACT_ELEMS / 4), nw4 = (int)(W_ELEMS / 4);
        dim3 ga((na4 + 255) / 256), gw((nw4 + 255) / 256), blk(256);
        cast_bf16_kernel<<<ga, blk, 0, stream>>>(q_in, qx, na4);
        cast_bf16_kernel<<<ga, blk, 0, stream>>>(k_in, kx, na4);
        cast_bf16_kernel<<<ga, blk, 0, stream>>>(v_in, vx, na4);
        cast_bf16_kernel<<<gw, blk, 0, stream>>>(Wq, wqx, nw4);
        cast_bf16_kernel<<<gw, blk, 0, stream>>>(Wk, wkx, nw4);
        cast_bf16_kernel<<<gw, blk, 0, stream>>>(Wv, wvx, nw4);
        cast_bf16_kernel<<<gw, blk, 0, stream>>>(Wo, wox, nw4);
    }

    // 2) Q/K/V projections (scale 1/sqrt(HD)=0.125 folded into Q)
    {
        dim3 g(MROWS / GBM, DMODEL / GBN), blk(256);
        gemm_xwT_wmma<true><<<g, blk, 0, stream>>>(qx, wqx, bq, (void*)Qp,
                                                   MROWS, DMODEL, DMODEL, 0.125f);
        gemm_xwT_wmma<true><<<g, blk, 0, stream>>>(kx, wkx, bk, (void*)Kp,
                                                   MROWS, DMODEL, DMODEL, 1.0f);
        gemm_xwT_wmma<true><<<g, blk, 0, stream>>>(vx, wvx, bv, (void*)Vp,
                                                   MROWS, DMODEL, DMODEL, 1.0f);
    }

    // 3) fused attention (needs >64KB dynamic LDS; CDNA5 WGP has 320KB)
    {
        static_assert(ATTN_LDS <= 320 * 1024, "LDS over WGP budget");
        (void)hipFuncSetAttribute((const void*)attn_fused_wmma,
                                  hipFuncAttributeMaxDynamicSharedMemorySize, ATTN_LDS);
        dim3 g(BATCH * NHEAD * (SEQ / QT)), blk(256);
        attn_fused_wmma<<<g, blk, ATTN_LDS, stream>>>(Qp, Kp, Vp, mask,
                                                      weights_out, Cx);
    }

    // 4) output projection -> fp32
    {
        dim3 g(MROWS / GBM, DMODEL / GBN), blk(256);
        gemm_xwT_wmma<false><<<g, blk, 0, stream>>>(Cx, wox, bo, (void*)out,
                                                    MROWS, DMODEL, DMODEL, 1.0f);
    }
}
